// EfficientChannelGroupAttention_9749575762712
// MI455X (gfx1250) — compile-verified, matrix operand order checked by reading, not run on hardware
//
#include <hip/hip_runtime.h>

typedef __bf16 bf16_t;
typedef __attribute__((ext_vector_type(16))) __bf16 v16bf;
typedef __attribute__((ext_vector_type(8)))  __bf16 v8bf;
typedef __attribute__((ext_vector_type(8)))  float  v8f;
typedef unsigned short u16;

#define DEVINL __device__ __forceinline__

namespace cfg {
constexpr int B = 8, N = 1024, C = 1024, G = 8, H = 4, BANK = 64;
constexpr int Cg = 128, Cc = 512, Cpg = 64, Hd = 16;
constexpr int M = N + BANK;   // 1088
}

// native f32 -> bf16 conversion (lets clang use the hardware cvt path)
DEVINL u16 f2bf(float f) { return __builtin_bit_cast(u16, (bf16_t)f); }
DEVINL bf16_t bfz() { return __builtin_bit_cast(bf16_t, (u16)0); }

DEVINL v8f wmma_bf16(v16bf a, v16bf b, v8f c) {
  // D(f32 16x16) = A(bf16 16x32) * B(bf16 32x16) + C
  return __builtin_amdgcn_wmma_f32_16x16x32_bf16(false, a, false, b, (short)0, c,
                                                 false, false);
}

// A/B fragment, full K=32.  A: row = lane&15; lanes<16 hold K {0..7,16..23},
// lanes>=16 hold K {8..15,24..31}.  B uses the same pattern on transposed data.
DEVINL v16bf load_frag_full(const bf16_t* base, int rstride) {
  int lane = threadIdx.x & 31;
  int ko = (lane >> 4) << 3;                 // 0 or 8
  const bf16_t* p = base + (lane & 15) * rstride + ko;
  v8bf lo = *(const v8bf*)p;                 // K = ko..ko+7      (16B load)
  v8bf hi = *(const v8bf*)(p + 16);          // K = 16+ko..16+ko+7
  v16bf f;
#pragma unroll
  for (int i = 0; i < 8; ++i) { f[i] = lo[i]; f[i + 8] = hi[i]; }
  return f;
}
// Fragment with only K=0..15 real (Hd=16 contraction), upper K zero-padded.
DEVINL v16bf load_frag_k16(const bf16_t* base, int rstride) {
  int lane = threadIdx.x & 31;
  int ko = (lane >> 4) << 3;
  const bf16_t* p = base + (lane & 15) * rstride + ko;
  v8bf lo = *(const v8bf*)p;
  v16bf f;
#pragma unroll
  for (int i = 0; i < 8; ++i) { f[i] = lo[i]; f[i + 8] = bfz(); }
  return f;
}

// reductions across a 16-lane half of the wave (rows live in lane halves)
DEVINL float half_max(float v) {
#pragma unroll
  for (int m = 1; m <= 8; m <<= 1) v = fmaxf(v, __shfl_xor(v, m, 32));
  return v;
}
DEVINL float half_sum(float v) {
#pragma unroll
  for (int m = 1; m <= 8; m <<= 1) v += __shfl_xor(v, m, 32);
  return v;
}

// ---------------------------------------------------------------------------
// Kernel 1: bank compression  kb = gK @ Wbk + bbk (64x64), vb likewise.
// kb stored row-major per head (H,BANK,Hd); vb stored transposed (H,Hd,BANK).
// ---------------------------------------------------------------------------
__global__ __launch_bounds__(256)
void ecga_bank_kernel(const float* __restrict__ gk, const float* __restrict__ gv,
                      const float* __restrict__ Wbk, const float* __restrict__ bbk,
                      const float* __restrict__ Wbv, const float* __restrict__ bbv,
                      u16* __restrict__ kbb, u16* __restrict__ vbt) {
  using namespace cfg;
  int idx = blockIdx.x * 256 + threadIdx.x;    // 0..8191
  int which = idx >> 12;                       // 0=K bank, 1=V bank
  int rem = idx & 4095;
  int m = rem >> 6;                            // bank row
  int col = rem & 63;                          // compressed channel
  const float* src  = which ? gv : gk;
  const float* W    = which ? Wbv : Wbk;
  const float* bias = which ? bbv : bbk;
  float acc = bias[col];
  const float* xr = src + (size_t)m * C;
#pragma unroll 4
  for (int c = 0; c < C; ++c) acc = fmaf(xr[c], W[(size_t)c * Cpg + col], acc);
  int h = col >> 4, d = col & 15;
  if (which) vbt[(size_t)h * Hd * BANK + d * BANK + m] = f2bf(acc);
  else       kbb[(size_t)h * BANK * Hd + m * Hd + d]   = f2bf(acc);
}

// ---------------------------------------------------------------------------
// Kernel 2: per-(b,g) QKV projection via WMMA.
// q,k -> (B,G,N,64) bf16 row-major;  v -> transposed (B,G,64,N) bf16.
// ---------------------------------------------------------------------------
__global__ __launch_bounds__(256)
void ecga_qkv_kernel(const float* __restrict__ x,
                     const float* __restrict__ Wq, const float* __restrict__ bq,
                     const float* __restrict__ Wk, const float* __restrict__ bk,
                     const float* __restrict__ Wv, const float* __restrict__ bv,
                     u16* __restrict__ qbuf, u16* __restrict__ kbuf,
                     u16* __restrict__ vtbuf) {
  using namespace cfg;
  extern __shared__ __align__(16) char smemraw[];
  bf16_t* xs  = (bf16_t*)smemraw;      // [128 rows][128 k]   32 KB
  bf16_t* wqT = xs + 128 * 128;        // [64 col][128 k]     16 KB each
  bf16_t* wkT = wqT + 64 * 128;
  bf16_t* wvT = wkT + 64 * 128;        // total 80 KB dynamic LDS

  int nb = blockIdx.x & 7;
  int g  = (blockIdx.x >> 3) & 7;
  int b  = blockIdx.x >> 6;
  int tid = threadIdx.x;
  int nrow0 = nb * 128;

#pragma unroll 4
  for (int i = 0; i < 64; ++i) {            // x tile f32 -> bf16 LDS
    int idx = i * 256 + tid;
    int r = idx >> 7, c = idx & 127;
    float vx = x[((size_t)b * N + nrow0 + r) * C + (size_t)g * Cg + c];
    ((u16*)xs)[r * 128 + c] = f2bf(vx);
  }
#pragma unroll 4
  for (int i = 0; i < 32; ++i) {            // weights, stored transposed
    int idx = i * 256 + tid;
    int k = idx >> 6, col = idx & 63;
    ((u16*)wqT)[col * 128 + k] = f2bf(Wq[(size_t)k * Cpg + col]);
    ((u16*)wkT)[col * 128 + k] = f2bf(Wk[(size_t)k * Cpg + col]);
    ((u16*)wvT)[col * 128 + k] = f2bf(Wv[(size_t)k * Cpg + col]);
  }
  __syncthreads();

  int w = tid >> 5, lane = tid & 31, half = lane >> 4;

  // 96 result tiles (8 row x 4 col x 3 matrices), 12 per wave.
  // Tiles grouped so 4 column-tiles share one row-tile: A fragments loaded
  // once per group and reused across 4 WMMAs each.
  for (int gi = 0; gi < 3; ++gi) {
    int tb  = w * 12 + gi * 4;        // group-aligned (multiple of 4)
    int mat = tb >> 5;                // 0=q 1=k 2=v (constant within group)
    int rem = tb & 31;
    int rt  = rem >> 2;               // row tile (constant within group)
    const bf16_t* wT = (mat == 0) ? wqT : (mat == 1) ? wkT : wvT;
    const float* bias = (mat == 0) ? bq : (mat == 1) ? bk : bv;

    v16bf a[4];
#pragma unroll
    for (int ks = 0; ks < 4; ++ks)
      a[ks] = load_frag_full(xs + rt * 16 * 128 + ks * 32, 128);

#pragma unroll
    for (int ct = 0; ct < 4; ++ct) {
      v8f acc = {};
#pragma unroll
      for (int ks = 0; ks < 4; ++ks) {  // K=128 in 4 WMMA steps
        v16bf bb = load_frag_full(wT + ct * 16 * 128 + ks * 32, 128);
        acc = wmma_bf16(a[ks], bb, acc);
      }
      int colg = ct * 16 + (lane & 15);
      float bcol = bias[colg];
      if (mat == 2) {                   // V stored transposed (dim-major)
        u16* dst = vtbuf + ((size_t)(b * G + g) * Cpg + colg) * N
                 + nrow0 + rt * 16 + half * 8;
#pragma unroll
        for (int r = 0; r < 8; ++r) dst[r] = f2bf(acc[r] + bcol);
      } else {
        u16* dst = ((mat == 0) ? qbuf : kbuf)
                 + ((size_t)(b * G + g) * N + nrow0 + rt * 16 + half * 8) * Cpg + colg;
#pragma unroll
        for (int r = 0; r < 8; ++r) dst[(size_t)r * Cpg] = f2bf(acc[r] + bcol);
      }
    }
  }
}

// ---------------------------------------------------------------------------
// Kernel 3: flash attention per (b,g,h).  Wave = 16 query rows, M streamed in
// chunks of 32 keys; token chunks and bank chunks in separate loops so the
// hot loop has no pointer selection.
// ---------------------------------------------------------------------------
__global__ __launch_bounds__(128)
void ecga_attn_kernel(const u16* __restrict__ qbuf_, const u16* __restrict__ kbuf_,
                      const u16* __restrict__ vtbuf_, const u16* __restrict__ kbb_,
                      const u16* __restrict__ vbt_, u16* __restrict__ attnbuf) {
  using namespace cfg;
  __shared__ __align__(16) u16 pshare[4][16 * 32];   // per-wave P transpose tile

  int nb = blockIdx.x & 15;
  int h  = (blockIdx.x >> 4) & 3;
  int g  = (blockIdx.x >> 6) & 7;
  int b  = blockIdx.x >> 9;
  int w = threadIdx.x >> 5, lane = threadIdx.x & 31, half = lane >> 4;
  int qrow0 = nb * 64 + w * 16;

  const bf16_t* qbg   = (const bf16_t*)qbuf_  + ((size_t)(b * G + g) * N) * Cpg + h * Hd;
  const bf16_t* kbg   = (const bf16_t*)kbuf_  + ((size_t)(b * G + g) * N) * Cpg + h * Hd;
  const bf16_t* vtg   = (const bf16_t*)vtbuf_ + ((size_t)(b * G + g) * Cpg + h * Hd) * N;
  const bf16_t* kbank = (const bf16_t*)kbb_ + (size_t)h * BANK * Hd;
  const bf16_t* vbank = (const bf16_t*)vbt_ + (size_t)h * Hd * BANK;
  bf16_t* pls = (bf16_t*)pshare[w];

  const float scale = 0.25f;  // 1/sqrt(Hd)

  v16bf qf = load_frag_k16(qbg + (size_t)qrow0 * Cpg, Cpg);

  float mrun[8], lrun[8];
  v8f acc = {};
#pragma unroll
  for (int r = 0; r < 8; ++r) { mrun[r] = -1e30f; lrun[r] = 0.0f; }

  auto process_chunk = [&](const bf16_t* kb, int kst, const bf16_t* vb, int vst) {
    // scores: q(16x16 padded to K=32) @ k^T -> two 16x16 tiles
    v16bf kf0 = load_frag_k16(kb, kst);
    v16bf kf1 = load_frag_k16(kb + 16 * kst, kst);
    v8f z = {};
    v8f s0 = wmma_bf16(qf, kf0, z);
    v8f s1 = wmma_bf16(qf, kf1, z);

    float p0[8], p1[8];
#pragma unroll
    for (int r = 0; r < 8; ++r) {     // online softmax; row = r + 8*half
      float a  = s0[r] * scale, c2 = s1[r] * scale;
      float cmax = half_max(fmaxf(a, c2));
      float mnew = fmaxf(mrun[r], cmax);
      float corr = __expf(mrun[r] - mnew);
      lrun[r] *= corr;
      acc[r] = acc[r] * corr;
      p0[r] = __expf(a - mnew);
      p1[r] = __expf(c2 - mnew);
      lrun[r] += half_sum(p0[r] + p1[r]);
      mrun[r] = mnew;
    }
    // transpose P (C-layout -> A-layout) through per-wave LDS tile
#pragma unroll
    for (int r = 0; r < 8; ++r) {
      int row = r + half * 8;
      ((u16*)pls)[row * 32 + (lane & 15)]      = f2bf(p0[r]);
      ((u16*)pls)[row * 32 + 16 + (lane & 15)] = f2bf(p1[r]);
    }
    v16bf pf = load_frag_full(pls, 32);             // in-order DS within wave
    v16bf vf = load_frag_full(vb, vst);             // V^T: 2x16B per lane
    acc = wmma_bf16(pf, vf, acc);
  };

  for (int m0 = 0; m0 < N; m0 += 32) {              // token keys (hot loop)
    __builtin_prefetch(kbg + (size_t)(m0 + 32) * Cpg, 0, 0);  // global_prefetch_b8
    process_chunk(kbg + (size_t)m0 * Cpg, Cpg, vtg + m0, N);
  }
#pragma unroll
  for (int m0 = 0; m0 < BANK; m0 += 32)             // global bank (2 chunks)
    process_chunk(kbank + (size_t)m0 * Hd, Hd, vbank + m0, BANK);

  u16* ob = attnbuf + ((size_t)(b * G + g) * N + qrow0 + half * 8) * Cpg
          + h * Hd + (lane & 15);
#pragma unroll
  for (int r = 0; r < 8; ++r) ob[(size_t)r * Cpg] = f2bf(acc[r] / lrun[r]);
}

// ---------------------------------------------------------------------------
// Kernel 4: output projection (B,N,512) @ Wo(512,1024) + bo -> f32 out.
// Block tile 128x64, K=512 in 16 WMMA steps.
// ---------------------------------------------------------------------------
__global__ __launch_bounds__(256)
void ecga_proj_kernel(const u16* __restrict__ attnbuf, const float* __restrict__ Wo,
                      const float* __restrict__ bo, float* __restrict__ out) {
  using namespace cfg;
  __shared__ __align__(16) u16 As[128 * 32];   // 8 KB
  __shared__ __align__(16) u16 Bs[64 * 32];    // 4 KB (Wo^T tile)

  int cb = blockIdx.x & 15;
  int nb = (blockIdx.x >> 4) & 7;
  int b  = blockIdx.x >> 7;
  int tid = threadIdx.x;
  int w = tid >> 5, lane = tid & 31, half = lane >> 4;
  int nrow0 = nb * 128, cb0 = cb * 64;

  v8f acc[4] = {{}, {}, {}, {}};

  for (int ks = 0; ks < 16; ++ks) {
    int k0 = ks * 32;
    __syncthreads();
#pragma unroll 4
    for (int i = 0; i < 16; ++i) {      // A: gather 128x32 from (B,G,N,64) bf16
      int idx = i * 256 + tid;
      int r = idx >> 5, kk = idx & 31;
      int gk2 = k0 + kk;
      int gg = gk2 >> 6, c2 = gk2 & 63;
      As[r * 32 + kk] = attnbuf[((size_t)(b * G + gg) * N + nrow0 + r) * Cpg + c2];
    }
#pragma unroll 4
    for (int i = 0; i < 8; ++i) {       // B: Wo tile transposed, f32 -> bf16
      int idx = i * 256 + tid;
      int col = idx >> 5, kk = idx & 31;
      Bs[col * 32 + kk] = f2bf(Wo[(size_t)(k0 + kk) * C + cb0 + col]);
    }
    __syncthreads();
    v16bf a = load_frag_full((const bf16_t*)As + w * 16 * 32, 32);
#pragma unroll
    for (int ct = 0; ct < 4; ++ct) {
      v16bf bb = load_frag_full((const bf16_t*)Bs + ct * 16 * 32, 32);
      acc[ct] = wmma_bf16(a, bb, acc[ct]);
    }
  }

  int row0 = nrow0 + w * 16 + half * 8;
#pragma unroll
  for (int ct = 0; ct < 4; ++ct) {
    int col = cb0 + ct * 16 + (lane & 15);
    float bb = bo[col];
#pragma unroll
    for (int r = 0; r < 8; ++r)
      out[((size_t)b * N + row0 + r) * C + col] = acc[ct][r] + bb;
  }
}

// ---------------------------------------------------------------------------
extern "C" void kernel_launch(void* const* d_in, const int* in_sizes, int n_in,
                              void* d_out, int out_size, void* d_ws, size_t ws_size,
                              hipStream_t stream) {
  using namespace cfg;
  const float* x   = (const float*)d_in[0];
  const float* Wq  = (const float*)d_in[1];
  const float* bq  = (const float*)d_in[2];
  const float* Wk  = (const float*)d_in[3];
  const float* bk  = (const float*)d_in[4];
  const float* Wv  = (const float*)d_in[5];
  const float* bv  = (const float*)d_in[6];
  const float* Wbk = (const float*)d_in[7];
  const float* bbk = (const float*)d_in[8];
  const float* Wbv = (const float*)d_in[9];
  const float* bbv = (const float*)d_in[10];
  const float* Wo  = (const float*)d_in[11];
  const float* bo  = (const float*)d_in[12];
  const float* gk  = (const float*)d_in[13];
  const float* gv  = (const float*)d_in[14];

  size_t qkElems = (size_t)B * G * N * Cpg;          // 4M bf16 each
  u16* qbuf    = (u16*)d_ws;
  u16* kbuf    = qbuf + qkElems;
  u16* vtbuf   = kbuf + qkElems;                     // V transposed (B,G,64,N)
  u16* attnbuf = vtbuf + qkElems;                    // (B,G,N,64)
  u16* kbb     = attnbuf + qkElems;                  // (H,BANK,Hd)
  u16* vbt     = kbb + (size_t)H * BANK * Hd;        // (H,Hd,BANK)

  ecga_bank_kernel<<<32, 256, 0, stream>>>(gk, gv, Wbk, bbk, Wbv, bbv, kbb, vbt);
  ecga_qkv_kernel<<<B * G * (N / 128), 256, 80 * 1024, stream>>>(
      x, Wq, bq, Wk, bk, Wv, bv, qbuf, kbuf, vtbuf);
  ecga_attn_kernel<<<B * G * H * (N / 64), 128, 0, stream>>>(
      qbuf, kbuf, vtbuf, kbb, vbt, attnbuf);
  ecga_proj_kernel<<<B * (N / 128) * (C / 64), 256, 0, stream>>>(
      attnbuf, Wo, bo, (float*)d_out);
}
